// QComplexSelfAttention_78254304133584
// MI455X (gfx1250) — compile-verified, matrix operand order checked by reading, not run on hardware
//
#include <hip/hip_runtime.h>
#include <cmath>

// ---------------------------------------------------------------------------
// CDNA5 (gfx1250) wave32 WMMA implementation of QComplexSelfAttention.
// All GEMMs use v_wmma_f32_16x16x32_bf16 (f32 accumulate), register-blocked
// 2x2 tiles (32x32) per wave: 16 WMMAs per 16 b128 loads per K-step.
// ---------------------------------------------------------------------------

typedef __attribute__((ext_vector_type(8)))  float  v8f;
typedef __attribute__((ext_vector_type(4)))  __bf16 bf16x4;
typedef __attribute__((ext_vector_type(8)))  __bf16 bf16x8;
typedef __attribute__((ext_vector_type(16))) __bf16 bf16x16;

// Load one 16x32 bf16 fragment (A-layout per CDNA5 ISA 7.12.2):
//   lane 0-15 : row = lane,    K = {0..7} in v0..3, {16..23} in v4..7
//   lane 16-31: row = lane-16, K = {8..15} in v0..3, {24..31} in v4..7
// Two aligned 16B loads per lane -> global_load_b128.
__device__ __forceinline__ bf16x16 load_frag16x32(const __bf16* __restrict__ p,
                                                  int ld, int lane) {
  const int row = lane & 15;
  const int kb  = (lane & 16) ? 8 : 0;
  const __bf16* r = p + (size_t)row * ld + kb;
  bf16x8 lo = *reinterpret_cast<const bf16x8*>(r);
  bf16x8 hi = *reinterpret_cast<const bf16x8*>(r + 16);
  bf16x16 f;
#pragma unroll
  for (int i = 0; i < 8; ++i) { f[i] = lo[i]; f[i + 8] = hi[i]; }
  return f;
}

__device__ __forceinline__ v8f wmma_bf16(bf16x16 a, bf16x16 b, v8f c) {
  return __builtin_amdgcn_wmma_f32_16x16x32_bf16(
      /*neg_a=*/false, a, /*neg_b=*/false, b,
      /*c_mod=*/(short)0, c, /*reuse_a=*/false, /*reuse_b=*/false);
}

// ---------------------------------------------------------------------------
// fp32 -> bf16 cast (vectorized, n multiple of 4)
// ---------------------------------------------------------------------------
__global__ __launch_bounds__(256)
void cast_f32_bf16_kernel(const float* __restrict__ src,
                          __bf16* __restrict__ dst, int n4) {
  int i = blockIdx.x * 256 + threadIdx.x;
  if (i < n4) {
    const float4 v = reinterpret_cast<const float4*>(src)[i];
    bf16x4 o;
    o[0] = (__bf16)v.x; o[1] = (__bf16)v.y;
    o[2] = (__bf16)v.z; o[3] = (__bf16)v.w;
    reinterpret_cast<bf16x4*>(dst)[i] = o;
  }
}

// ---------------------------------------------------------------------------
// Complex dense:  Pa = Xa Wa^T + j2 * Xb Wb^T + ba ;  Pb = Xa Wb^T + Xb Wa^T + bb
// X: [M,K] row-major, W: [N,K] row-major (B-fragments read W rows).
// Block = 8 waves (2 m x 4 n) -> 64x128 tile; wave = 2x2 16x16 tiles.
// WRITE_F32 selects the output path at compile time so the epilogue is
// branch-free and fully unrolled (no per-element pointer tests).
// ---------------------------------------------------------------------------
template <bool WRITE_F32>
__global__ __launch_bounds__(256)
void qdense_kernel(const __bf16* __restrict__ Xa, const __bf16* __restrict__ Xb,
                   const __bf16* __restrict__ Wa, const __bf16* __restrict__ Wb,
                   const float* __restrict__ bias_a, const float* __restrict__ bias_b,
                   __bf16* __restrict__ Pa_bf, __bf16* __restrict__ Pb_bf,
                   float* __restrict__ Pa_f, float* __restrict__ Pb_f,
                   int N, int K, float j2) {
  const int lane = threadIdx.x & 31;
  const int wave = threadIdx.x >> 5;
  const int m0 = blockIdx.x * 64  + (wave & 1) * 32;   // 2 waves in M
  const int n0 = blockIdx.y * 128 + (wave >> 1) * 32;  // 4 waves in N

  const __bf16* xa0 = Xa + (size_t)m0 * K;
  const __bf16* xb0 = Xb + (size_t)m0 * K;
  const __bf16* wa0 = Wa + (size_t)n0 * K;
  const __bf16* wb0 = Wb + (size_t)n0 * K;

  v8f acc_aa[2][2] = {}, acc_bb[2][2] = {}, acc_ab[2][2] = {}, acc_ba[2][2] = {};

  for (int k = 0; k < K; k += 32) {
    bf16x16 fxa[2], fxb[2], fwa[2], fwb[2];
#pragma unroll
    for (int mi = 0; mi < 2; ++mi) {
      fxa[mi] = load_frag16x32(xa0 + (size_t)(mi * 16) * K + k, K, lane);
      fxb[mi] = load_frag16x32(xb0 + (size_t)(mi * 16) * K + k, K, lane);
    }
#pragma unroll
    for (int ni = 0; ni < 2; ++ni) {
      fwa[ni] = load_frag16x32(wa0 + (size_t)(ni * 16) * K + k, K, lane);
      fwb[ni] = load_frag16x32(wb0 + (size_t)(ni * 16) * K + k, K, lane);
    }
#pragma unroll
    for (int mi = 0; mi < 2; ++mi)
#pragma unroll
      for (int ni = 0; ni < 2; ++ni) {
        acc_aa[mi][ni] = wmma_bf16(fxa[mi], fwa[ni], acc_aa[mi][ni]);
        acc_bb[mi][ni] = wmma_bf16(fxb[mi], fwb[ni], acc_bb[mi][ni]);
        acc_ab[mi][ni] = wmma_bf16(fxa[mi], fwb[ni], acc_ab[mi][ni]);
        acc_ba[mi][ni] = wmma_bf16(fxb[mi], fwa[ni], acc_ba[mi][ni]);
      }
  }

  // C/D layout: lane holds N = lane&15; VGPR r holds M = r (+8 for hi lanes)
  const int ncol  = lane & 15;
  const int mhalf = (lane & 16) ? 8 : 0;
  float biasA[2], biasB[2];
#pragma unroll
  for (int ni = 0; ni < 2; ++ni) {
    biasA[ni] = bias_a[n0 + ni * 16 + ncol];
    biasB[ni] = bias_b[n0 + ni * 16 + ncol];
  }
#pragma unroll
  for (int ni = 0; ni < 2; ++ni) {
    const int n = n0 + ni * 16 + ncol;
#pragma unroll
    for (int mi = 0; mi < 2; ++mi) {
      const size_t base = (size_t)(m0 + mi * 16 + mhalf) * N + n;
#pragma unroll
      for (int r = 0; r < 8; ++r) {
        const size_t idx = base + (size_t)r * N;
        const float pa = acc_aa[mi][ni][r] + j2 * acc_bb[mi][ni][r] + biasA[ni];
        const float pb = acc_ab[mi][ni][r] + acc_ba[mi][ni][r] + biasB[ni];
        if constexpr (WRITE_F32) {
          Pa_f[idx] = pa;
          Pb_f[idx] = pb;
        } else {
          Pa_bf[idx] = (__bf16)pa;
          Pb_bf[idx] = (__bf16)pb;
        }
      }
    }
  }
}

// ---------------------------------------------------------------------------
// scores[b,s,t] = sqrt(sa^2 + sb^2) / sqrt(D)
//   sa = QaKa - c2*QbKb ; sb = QaKb + QbKa   (dot products over D)
// GEMM M=S, N=S, K=D; 2x2 register blocking like qdense.
// ---------------------------------------------------------------------------
__global__ __launch_bounds__(256)
void scores_kernel(const __bf16* __restrict__ Qa, const __bf16* __restrict__ Qb,
                   const __bf16* __restrict__ Ka, const __bf16* __restrict__ Kb,
                   float* __restrict__ scores, int S, int Dd,
                   float c2, float invSqrtD) {
  const int lane = threadIdx.x & 31;
  const int wave = threadIdx.x >> 5;
  const int b = blockIdx.z;
  const size_t boff = (size_t)b * S * Dd;
  const int s0 = blockIdx.x * 64  + (wave & 1) * 32;
  const int t0 = blockIdx.y * 128 + (wave >> 1) * 32;

  const __bf16* qa0 = Qa + boff + (size_t)s0 * Dd;
  const __bf16* qb0 = Qb + boff + (size_t)s0 * Dd;
  const __bf16* ka0 = Ka + boff + (size_t)t0 * Dd;
  const __bf16* kb0 = Kb + boff + (size_t)t0 * Dd;

  v8f acc_aa[2][2] = {}, acc_bb[2][2] = {}, acc_ab[2][2] = {}, acc_ba[2][2] = {};

  for (int k = 0; k < Dd; k += 32) {
    bf16x16 fqa[2], fqb[2], fka[2], fkb[2];
#pragma unroll
    for (int mi = 0; mi < 2; ++mi) {
      fqa[mi] = load_frag16x32(qa0 + (size_t)(mi * 16) * Dd + k, Dd, lane);
      fqb[mi] = load_frag16x32(qb0 + (size_t)(mi * 16) * Dd + k, Dd, lane);
    }
#pragma unroll
    for (int ni = 0; ni < 2; ++ni) {
      fka[ni] = load_frag16x32(ka0 + (size_t)(ni * 16) * Dd + k, Dd, lane);
      fkb[ni] = load_frag16x32(kb0 + (size_t)(ni * 16) * Dd + k, Dd, lane);
    }
#pragma unroll
    for (int mi = 0; mi < 2; ++mi)
#pragma unroll
      for (int ni = 0; ni < 2; ++ni) {
        acc_aa[mi][ni] = wmma_bf16(fqa[mi], fka[ni], acc_aa[mi][ni]);
        acc_bb[mi][ni] = wmma_bf16(fqb[mi], fkb[ni], acc_bb[mi][ni]);
        acc_ab[mi][ni] = wmma_bf16(fqa[mi], fkb[ni], acc_ab[mi][ni]);
        acc_ba[mi][ni] = wmma_bf16(fqb[mi], fka[ni], acc_ba[mi][ni]);
      }
  }

  float* out = scores + (size_t)b * S * S;
  const int ncol  = lane & 15;
  const int mhalf = (lane & 16) ? 8 : 0;
#pragma unroll
  for (int ni = 0; ni < 2; ++ni) {
    const int t = t0 + ni * 16 + ncol;
#pragma unroll
    for (int mi = 0; mi < 2; ++mi) {
      const size_t base = (size_t)(s0 + mi * 16 + mhalf) * S + t;
#pragma unroll
      for (int r = 0; r < 8; ++r) {
        const float sa = acc_aa[mi][ni][r] - c2 * acc_bb[mi][ni][r];
        const float sv = acc_ab[mi][ni][r] + acc_ba[mi][ni][r];
        out[base + (size_t)r * S] = sqrtf(sa * sa + sv * sv) * invSqrtD;
      }
    }
  }
}

// ---------------------------------------------------------------------------
// Row softmax over S=2048 (one 256-thread block per row), f32 -> bf16
// ---------------------------------------------------------------------------
__global__ __launch_bounds__(256)
void softmax_kernel(const float* __restrict__ scores,
                    __bf16* __restrict__ attn, int S) {
  const size_t row = blockIdx.x;
  const float* src = scores + row * S;
  __bf16* dst = attn + row * S;
  const int lane = threadIdx.x & 31;
  const int wv   = threadIdx.x >> 5;
  __shared__ float smax[8];
  __shared__ float ssum[8];

  float loc[8];
  float mx = -3.4e38f;
#pragma unroll
  for (int i = 0; i < 8; ++i) {
    loc[i] = src[threadIdx.x + (i << 8)];
    mx = fmaxf(mx, loc[i]);
  }
#pragma unroll
  for (int o = 16; o > 0; o >>= 1) mx = fmaxf(mx, __shfl_xor(mx, o, 32));
  if (lane == 0) smax[wv] = mx;
  __syncthreads();
  mx = smax[0];
#pragma unroll
  for (int w = 1; w < 8; ++w) mx = fmaxf(mx, smax[w]);

  float sum = 0.f;
#pragma unroll
  for (int i = 0; i < 8; ++i) { loc[i] = __expf(loc[i] - mx); sum += loc[i]; }
#pragma unroll
  for (int o = 16; o > 0; o >>= 1) sum += __shfl_xor(sum, o, 32);
  if (lane == 0) ssum[wv] = sum;
  __syncthreads();
  sum = ssum[0];
#pragma unroll
  for (int w = 1; w < 8; ++w) sum += ssum[w];

  const float inv = 1.0f / sum;
#pragma unroll
  for (int i = 0; i < 8; ++i)
    dst[threadIdx.x + (i << 8)] = (__bf16)(loc[i] * inv);
}

// ---------------------------------------------------------------------------
// Per-batch 32x32 LDS-tiled transpose: Vt[b][d][s] = V[b*S+s][d]
// ---------------------------------------------------------------------------
__global__ __launch_bounds__(256)
void transpose_kernel(const __bf16* __restrict__ V, __bf16* __restrict__ Vt,
                      int S, int Dd) {
  __shared__ __bf16 tile[32][33];
  const int b = blockIdx.z;
  const __bf16* src = V + (size_t)b * S * Dd;
  __bf16* dst = Vt + (size_t)b * Dd * S;
  const int tx = threadIdx.x & 31;
  const int ty = threadIdx.x >> 5;
  const int s0 = blockIdx.x * 32;
  const int d0 = blockIdx.y * 32;
#pragma unroll
  for (int i = 0; i < 32; i += 8)
    tile[ty + i][tx] = src[(size_t)(s0 + ty + i) * Dd + d0 + tx];
  __syncthreads();
#pragma unroll
  for (int i = 0; i < 32; i += 8)
    dst[(size_t)(d0 + ty + i) * S + s0 + tx] = tile[tx][ty + i];
}

// ---------------------------------------------------------------------------
// Oa = attn @ Va, Ob = attn @ Vb.  GEMM M=S, N=D, K=S, B from transposed V.
// 2x2 register blocking: per K-step 6 fragments (12 loads) feed 8 WMMAs.
// ---------------------------------------------------------------------------
__global__ __launch_bounds__(256)
void attnv_kernel(const __bf16* __restrict__ attn,
                  const __bf16* __restrict__ Vta, const __bf16* __restrict__ Vtb,
                  __bf16* __restrict__ Oa, __bf16* __restrict__ Ob,
                  int S, int Dd) {
  const int lane = threadIdx.x & 31;
  const int wave = threadIdx.x >> 5;
  const int b = blockIdx.z;
  const int s0 = blockIdx.x * 64  + (wave & 1) * 32;
  const int d0 = blockIdx.y * 128 + (wave >> 1) * 32;

  const __bf16* at0 = attn + (size_t)b * S * S  + (size_t)s0 * S;
  const __bf16* va0 = Vta  + (size_t)b * Dd * S + (size_t)d0 * S;
  const __bf16* vb0 = Vtb  + (size_t)b * Dd * S + (size_t)d0 * S;

  v8f acc_a[2][2] = {}, acc_b[2][2] = {};

  for (int k = 0; k < S; k += 32) {
    bf16x16 fat[2], fva[2], fvb[2];
#pragma unroll
    for (int mi = 0; mi < 2; ++mi)
      fat[mi] = load_frag16x32(at0 + (size_t)(mi * 16) * S + k, S, lane);
#pragma unroll
    for (int ni = 0; ni < 2; ++ni) {
      fva[ni] = load_frag16x32(va0 + (size_t)(ni * 16) * S + k, S, lane);
      fvb[ni] = load_frag16x32(vb0 + (size_t)(ni * 16) * S + k, S, lane);
    }
#pragma unroll
    for (int mi = 0; mi < 2; ++mi)
#pragma unroll
      for (int ni = 0; ni < 2; ++ni) {
        acc_a[mi][ni] = wmma_bf16(fat[mi], fva[ni], acc_a[mi][ni]);
        acc_b[mi][ni] = wmma_bf16(fat[mi], fvb[ni], acc_b[mi][ni]);
      }
  }

  const int ncol  = lane & 15;
  const int mhalf = (lane & 16) ? 8 : 0;
#pragma unroll
  for (int ni = 0; ni < 2; ++ni) {
    const int d = d0 + ni * 16 + ncol;
#pragma unroll
    for (int mi = 0; mi < 2; ++mi) {
      const size_t base = (size_t)(b * S + s0 + mi * 16 + mhalf) * Dd + d;
#pragma unroll
      for (int r = 0; r < 8; ++r) {
        const size_t idx = base + (size_t)r * Dd;
        Oa[idx] = (__bf16)acc_a[mi][ni][r];
        Ob[idx] = (__bf16)acc_b[mi][ni][r];
      }
    }
  }
}

// ---------------------------------------------------------------------------
// Host orchestration.  Inputs (setup_inputs order):
//  0 xa, 1 xb, 2 q_wa, 3 q_wb, 4 q_ba, 5 q_bb, 6 k_wa, 7 k_wb, 8 k_ba, 9 k_bb,
// 10 v_wa, 11 v_wb, 12 v_ba, 13 v_bb, 14 o_wa, 15 o_wb, 16 o_ba, 17 o_bb
// d_out = [out_a | out_b] f32, each B*S*D.  Needs ~305 MB workspace.
// ---------------------------------------------------------------------------
extern "C" void kernel_launch(void* const* d_in, const int* in_sizes, int n_in,
                              void* d_out, int out_size, void* d_ws, size_t ws_size,
                              hipStream_t stream) {
  (void)in_sizes; (void)n_in; (void)out_size; (void)ws_size;
  constexpr int B = 4, S = 2048, D = 1024;
  constexpr int M = B * S;
  const size_t MD  = (size_t)M * D;
  const size_t DD  = (size_t)D * D;
  const size_t SSB = (size_t)B * S * S;

  const float c2 = cosf(2.0f);      // cos(2) used in scores
  const float j2 = -c2;             // j^2 = -cos(2*theta), theta = 1
  const float invSqrtD = 1.0f / sqrtf((float)D);

  char* ws = (char*)d_ws;
  auto carve = [&](size_t bytes) -> char* {
    char* p = ws;
    ws += (bytes + 255) & ~(size_t)255;
    return p;
  };

  __bf16* xa_bf = (__bf16*)carve(MD * 2);
  __bf16* xb_bf = (__bf16*)carve(MD * 2);
  __bf16* wbf[8];
  for (int i = 0; i < 8; ++i) wbf[i] = (__bf16*)carve(DD * 2);
  __bf16* Qa  = (__bf16*)carve(MD * 2);
  __bf16* Qb  = (__bf16*)carve(MD * 2);
  __bf16* Kka = (__bf16*)carve(MD * 2);
  __bf16* Kkb = (__bf16*)carve(MD * 2);
  __bf16* Va  = (__bf16*)carve(MD * 2);
  __bf16* Vb  = (__bf16*)carve(MD * 2);
  __bf16* Vta = (__bf16*)carve(MD * 2);
  __bf16* Vtb = (__bf16*)carve(MD * 2);
  float*  sc  = (float*) carve(SSB * 4);
  __bf16* at  = (__bf16*)carve(SSB * 2);
  __bf16* Oa  = (__bf16*)carve(MD * 2);
  __bf16* Ob  = (__bf16*)carve(MD * 2);

  auto cast_launch = [&](const void* src, __bf16* dst, size_t n) {
    const int n4 = (int)(n / 4);
    cast_f32_bf16_kernel<<<(n4 + 255) / 256, 256, 0, stream>>>(
        (const float*)src, dst, n4);
  };
  cast_launch(d_in[0], xa_bf, MD);
  cast_launch(d_in[1], xb_bf, MD);
  const int widx[8] = {2, 3, 6, 7, 10, 11, 14, 15};
  for (int i = 0; i < 8; ++i) cast_launch(d_in[widx[i]], wbf[i], DD);

  const dim3 gProj(M / 64, D / 128);   // 128 x 8 blocks, 256 threads
  qdense_kernel<false><<<gProj, 256, 0, stream>>>(
      xa_bf, xb_bf, wbf[0], wbf[1],
      (const float*)d_in[4], (const float*)d_in[5],
      Qa, Qb, nullptr, nullptr, D, D, j2);
  qdense_kernel<false><<<gProj, 256, 0, stream>>>(
      xa_bf, xb_bf, wbf[2], wbf[3],
      (const float*)d_in[8], (const float*)d_in[9],
      Kka, Kkb, nullptr, nullptr, D, D, j2);
  qdense_kernel<false><<<gProj, 256, 0, stream>>>(
      xa_bf, xb_bf, wbf[4], wbf[5],
      (const float*)d_in[12], (const float*)d_in[13],
      Va, Vb, nullptr, nullptr, D, D, j2);

  const dim3 gT(S / 32, D / 32, B);
  transpose_kernel<<<gT, 256, 0, stream>>>(Va, Vta, S, D);
  transpose_kernel<<<gT, 256, 0, stream>>>(Vb, Vtb, S, D);

  const dim3 gScores(S / 64, S / 128, B);
  scores_kernel<<<gScores, 256, 0, stream>>>(Qa, Qb, Kka, Kkb, sc, S, D,
                                             c2, invSqrtD);

  softmax_kernel<<<B * S, 256, 0, stream>>>(sc, at, S);

  const dim3 gAV(S / 64, D / 128, B);
  attnv_kernel<<<gAV, 256, 0, stream>>>(at, Vta, Vtb, Oa, Ob, S, D);

  float* out_a = (float*)d_out;
  float* out_b = out_a + MD;
  qdense_kernel<true><<<gProj, 256, 0, stream>>>(
      Oa, Ob, wbf[6], wbf[7],
      (const float*)d_in[16], (const float*)d_in[17],
      nullptr, nullptr, out_a, out_b, D, D, j2);
}
